// PointerAttentionMasking_9594956939691
// MI455X (gfx1250) — compile-verified, matrix operand order checked by reading, not run on hardware
//
#include <hip/hip_runtime.h>
#include <hip/hip_bf16.h>

// Problem constants (match reference)
static constexpr int B = 128, T = 512, L = 512, H = 512, U = 512;
static constexpr float BIG = 1000000.0f;

typedef __attribute__((ext_vector_type(16))) _Float16 v16h;
typedef __attribute__((ext_vector_type(8)))  float    v8f;

// ---------------------------------------------------------------------------
// CDNA5 hardware tanh (v_tanh_f32).
// ---------------------------------------------------------------------------
__device__ __forceinline__ float hw_tanh(float x) {
#if __has_builtin(__builtin_amdgcn_tanhf)
  return __builtin_amdgcn_tanhf(x);
#elif __has_builtin(__builtin_amdgcn_tanh_f32)
  return __builtin_amdgcn_tanh_f32(x);
#else
  float y;
  asm volatile("v_tanh_f32 %0, %1" : "=v"(y) : "v"(x));
  return y;
#endif
}

// ---------------------------------------------------------------------------
// CDNA5 async global->LDS copy (ASYNCcnt-tracked). The clang builtin takes
// addrspace(1) int* (printed as "__device__ int *") and addrspace(3) int*.
// ---------------------------------------------------------------------------
#if __has_builtin(__builtin_amdgcn_global_load_async_to_lds_b32)
#define HAVE_ASYNC_LDS 1
typedef __attribute__((address_space(1))) int gint_as;
typedef __attribute__((address_space(3))) int lint_as;
__device__ __forceinline__ void async_copy_b32(float* lds_dst, const float* gsrc) {
  __builtin_amdgcn_global_load_async_to_lds_b32(
      (gint_as*)gsrc, (lint_as*)lds_dst, /*offset=*/0, /*cpol=*/0);
}
__device__ __forceinline__ void wait_async0() {
#if __has_builtin(__builtin_amdgcn_s_wait_asynccnt)
  __builtin_amdgcn_s_wait_asynccnt(0);
#else
  asm volatile("s_wait_asynccnt 0x0" ::: "memory");
#endif
}
#endif

// ---------------------------------------------------------------------------
// Kernel 1: Y[M,U] = X[M,H] @ W[H,U] + bias, fp32 I/O, f16 WMMA compute.
// One wave per 16x16 tile; fragments per the ISA 16-bit 16x32 VGPR layout.
// ---------------------------------------------------------------------------
__global__ void proj_gemm_f16(const float* __restrict__ X,
                              const float* __restrict__ W,
                              const float* __restrict__ bias,
                              float* __restrict__ Y, int M) {
  const int lane   = threadIdx.x & 31;
  const int wave   = threadIdx.x >> 5;              // 0..7
  const int n0     = (blockIdx.x * 8 + wave) * 16;  // N tile base (U dim)
  const int m0     = blockIdx.y * 16;               // M tile base
  const int half16 = (lane >= 16) ? 8 : 0;
  const int lm     = lane & 15;
  const size_t rowA = (size_t)(m0 + lm) * H;

  v8f c = {};
  for (int kb = 0; kb < H; kb += 32) {
    v16h a, bfrag;
#pragma unroll
    for (int i = 0; i < 16; ++i) {
      const int v = i >> 1, p = i & 1;
      const int k = kb + ((v >> 2) << 4) + half16 + ((v & 3) << 1) + p;
      a[i]     = (_Float16)X[rowA + k];                 // A(m, k)
      bfrag[i] = (_Float16)W[(size_t)k * U + n0 + lm];  // B(k, n)
    }
    c = __builtin_amdgcn_wmma_f32_16x16x32_f16(
        false, a, false, bfrag, (short)0, c, false, false);
  }

  const float bv = bias[n0 + lm];
#pragma unroll
  for (int r = 0; r < 8; ++r) {                    // D: reg r -> M = r + half16
    const int m = m0 + r + half16;
    Y[(size_t)m * U + n0 + lm] = c[r] + bv;
  }
}

// ---------------------------------------------------------------------------
// Kernel 2: S[b,t,l] = bV + sum_u V[u] * tanh(dp[b,t,u] + ep[b,l,u]).
// 64x64 (t,l) tile / 256 threads, 4x4 register blocking, U chunked by 64
// through LDS (async global->LDS fills). Row stride 65 => conflict-free
// column gathers on ep_s. 9 LDS reads feed 16 tanh+FMA per u-step.
// ---------------------------------------------------------------------------
#define UCH 64
__global__ void score_kernel(const float* __restrict__ dp,
                             const float* __restrict__ ep,
                             const float* __restrict__ V,
                             const float* __restrict__ bV,
                             float* __restrict__ S) {
  __shared__ float dp_s[64][UCH + 1];
  __shared__ float ep_s[64][UCH + 1];
  __shared__ float Vs[U];

  const int b   = blockIdx.z;
  const int t0  = blockIdx.y * 64;
  const int l0  = blockIdx.x * 64;
  const int tid = threadIdx.x;                     // 256 threads

  for (int i = tid; i < U; i += 256) Vs[i] = V[i];

  const int tr = tid >> 4;                         // 0..15 -> t rows 4*tr..4*tr+3
  const int tc = tid & 15;                         // 0..15 -> l cols 4*tc..4*tc+3
  float acc[4][4] = {};

  const float* dpb = dp + ((size_t)b * T + t0) * U;
  const float* epb = ep + ((size_t)b * L + l0) * U;

  for (int uc = 0; uc < U; uc += UCH) {
    __syncthreads();                               // protect prior readers
    for (int i = tid; i < 64 * UCH; i += 256) {
      const int r = i >> 6, cc = i & 63;
#ifdef HAVE_ASYNC_LDS
      async_copy_b32(&dp_s[r][cc], dpb + (size_t)r * U + uc + cc);
      async_copy_b32(&ep_s[r][cc], epb + (size_t)r * U + uc + cc);
#else
      dp_s[r][cc] = dpb[(size_t)r * U + uc + cc];
      ep_s[r][cc] = epb[(size_t)r * U + uc + cc];
#endif
    }
#ifdef HAVE_ASYNC_LDS
    wait_async0();                                 // this wave's copies landed
#endif
    __syncthreads();                               // all waves' copies visible

    for (int u = 0; u < UCH; ++u) {
      const float vv = Vs[uc + u];
      float d[4], e[4];
#pragma unroll
      for (int i = 0; i < 4; ++i) d[i] = dp_s[4 * tr + i][u];
#pragma unroll
      for (int j = 0; j < 4; ++j) e[j] = ep_s[4 * tc + j][u];
#pragma unroll
      for (int i = 0; i < 4; ++i)
#pragma unroll
        for (int j = 0; j < 4; ++j)
          acc[i][j] += vv * hw_tanh(d[i] + e[j]);
    }
  }

  const float bv = bV[0];
  float* Sb = S + (size_t)b * T * L;
#pragma unroll
  for (int i = 0; i < 4; ++i)
#pragma unroll
    for (int j = 0; j < 4; ++j)
      Sb[(size_t)(t0 + 4 * tr + i) * L + (l0 + 4 * tc + j)] = acc[i][j] + bv;
}

// ---------------------------------------------------------------------------
// Kernel 3: sequential masked softmax + argmax scan, in place over S.
// One block per batch, 512 threads (16 wave32s), 2 barriers per step.
// ---------------------------------------------------------------------------
__global__ void scan_kernel(float* __restrict__ S) {
  __shared__ float wv[16];
  __shared__ int   wi[16];
  __shared__ float ws[16];

  const int b    = blockIdx.x;
  const int l    = threadIdx.x;                    // 0..511
  const int lane = l & 31;
  const int wid  = l >> 5;                         // 0..15
  float mask = 0.0f;
  float* Sb = S + (size_t)b * T * L;

  for (int t = 0; t < T; ++t) {
    const float s = Sb[(size_t)t * L + l] - mask * BIG;

    // ---- wave argmax (first-max tie-break: smaller index wins) ----
    float v = s; int idx = l;
#pragma unroll
    for (int m = 16; m > 0; m >>= 1) {
      const float v2 = __shfl_xor(v, m, 32);
      const int   i2 = __shfl_xor(idx, m, 32);
      if (v2 > v || (v2 == v && i2 < idx)) { v = v2; idx = i2; }
    }
    if (lane == 0) { wv[wid] = v; wi[wid] = idx; }
    __syncthreads();

    float mx = wv[0]; int elem = wi[0];
#pragma unroll
    for (int w = 1; w < 16; ++w) {
      const float v2 = wv[w];
      if (v2 > mx) { mx = v2; elem = wi[w]; }      // lower-w partials cover lower l
    }

    // ---- softmax ----
    const float e = __expf(s - mx);
    float ps = e;
#pragma unroll
    for (int m = 16; m > 0; m >>= 1) ps += __shfl_xor(ps, m, 32);
    if (lane == 0) ws[wid] = ps;
    __syncthreads();

    float sum = 0.0f;
#pragma unroll
    for (int w = 0; w < 16; ++w) sum += ws[w];

    Sb[(size_t)t * L + l] = e / sum;
    if (l == elem) mask += 1.0f;                   // one_hot accumulate
  }
}

// ---------------------------------------------------------------------------
extern "C" void kernel_launch(void* const* d_in, const int* in_sizes, int n_in,
                              void* d_out, int out_size, void* d_ws, size_t ws_size,
                              hipStream_t stream) {
  const float* dec = (const float*)d_in[0];  // [B,T,H]
  const float* enc = (const float*)d_in[1];  // [B,L,H]
  // d_in[2] (dec_input) only defines the mask shape; unused.
  const float* W1  = (const float*)d_in[3];  // [H,U]
  const float* b1  = (const float*)d_in[4];  // [U]
  const float* W2  = (const float*)d_in[5];  // [H,U]
  const float* b2  = (const float*)d_in[6];  // [U]
  const float* V   = (const float*)d_in[7];  // [U,1]
  const float* bV  = (const float*)d_in[8];  // [1]
  float* out = (float*)d_out;                // [B,T,L], reused as score scratch

  float* dp = (float*)d_ws;                  // [B*T, U] fp32
  float* ep = dp + (size_t)B * T * U;        // [B*L, U] fp32

  // 1) Projections via WMMA f16 (fp32 accumulate)
  proj_gemm_f16<<<dim3(U / 128, (B * T) / 16), 256, 0, stream>>>(dec, W1, b1, dp, B * T);
  proj_gemm_f16<<<dim3(U / 128, (B * L) / 16), 256, 0, stream>>>(enc, W2, b2, ep, B * L);

  // 2) Dense score tensor (tanh-bound), written into d_out
  score_kernel<<<dim3(L / 64, T / 64, B), 256, 0, stream>>>(dp, ep, V, bV, out);

  // 3) In-place masked softmax + pointer scan
  scan_kernel<<<B, L, 0, stream>>>(out);
}
